// graph_encode_26955214750315
// MI455X (gfx1250) — matrix operand order repaired and checked by reading, not compiled, not run on hardware
//
#include <hip/hip_runtime.h>
#include <hip/hip_bf16.h>
#include <math.h>

// Problem constants
#define BB   8
#define NN   1600
#define EE   800
#define RR   800
#define HSZ  128
#define NH   4
#define DKD  32
#define MROWS (BB*NN)     // 12800
#define NW   (NN/32)      // 50 packed mask words per row

typedef __attribute__((ext_vector_type(16))) __bf16 v16bf;
typedef __attribute__((ext_vector_type(8)))  float  v8f;

union FragCast { uint4 q[2]; v16bf v; };

// A-fragment: 16x32 bf16, storage row-major [m][k] with 'stride' elements per row.
static __device__ __forceinline__ v16bf load_frag_a(const __bf16* base, int stride) {
  int lane = threadIdx.x & 31;
  int m  = lane & 15;
  int kh = lane >> 4;
  const __bf16* p = base + (size_t)m * stride + kh * 8;
  FragCast f;
  f.q[0] = *(const uint4*)(p);
  f.q[1] = *(const uint4*)(p + 16);
  return f.v;
}

// B-fragment: 32x16 bf16 (KxN), storage TRANSPOSED [n][k] with 'stride' elems per row.
static __device__ __forceinline__ v16bf load_frag_b(const __bf16* base, int stride) {
  int lane = threadIdx.x & 31;
  int n  = lane & 15;
  int kh = lane >> 4;
  const __bf16* p = base + (size_t)n * stride + kh * 8;
  FragCast f;
  f.q[0] = *(const uint4*)(p);
  f.q[1] = *(const uint4*)(p + 16);
  return f.v;
}

static __device__ __forceinline__ v8f wmma_bf16(v16bf a, v16bf b, v8f c) {
  return __builtin_amdgcn_wmma_f32_16x16x32_bf16(false, a, false, b, (short)0, c, false, false);
}

// ---- CDNA5 async global->LDS staging (ASYNCcnt tracked, no VGPR data) ----
static __device__ __forceinline__ uint32_t lds_offset_of(const void* p) {
  // generic LDS pointers carry the LDS byte address in the low 32 bits
  return (uint32_t)(uintptr_t)p;
}

// copy 32KB contiguous: 256 threads x 16B x 8 iterations
static __device__ __forceinline__ void async_copy_contig(uint32_t lds_off, const void* gsrc) {
  uint64_t gb = (uint64_t)(uintptr_t)gsrc;
  uint32_t off = (uint32_t)threadIdx.x * 16u;
#pragma unroll
  for (int i = 0; i < 8; ++i) {
    asm volatile("global_load_async_to_lds_b128 %0, %1, %2"
                 :: "v"(lds_off + off), "v"(off), "s"(gb)
                 : "memory");
    off += 4096u;
  }
}

// copy 128 rows x 256B from strided global rows into contiguous LDS [128][128] bf16
static __device__ __forceinline__ void async_copy_rows(uint32_t lds_off, const void* gsrc,
                                                       uint32_t grow_stride_bytes) {
  uint64_t gb = (uint64_t)(uintptr_t)gsrc;
#pragma unroll
  for (int i = 0; i < 8; ++i) {
    uint32_t idx = (uint32_t)threadIdx.x + (uint32_t)i*256u;   // 16B chunk id, 0..2047
    uint32_t r = idx >> 4, c = idx & 15u;
    uint32_t goff = r * grow_stride_bytes + c*16u;
    asm volatile("global_load_async_to_lds_b128 %0, %1, %2"
                 :: "v"(lds_off + idx*16u), "v"(goff), "s"(gb)
                 : "memory");
  }
}

static __device__ __forceinline__ void async_wait_all() {
  asm volatile("s_wait_asynccnt 0x0" ::: "memory");
}

// ---------------------------------------------------------------- adjacency bit-pack
// 20.48M entries -> 2.56MB of bits (L2-resident for all 8 attention passes)
__global__ void pack_adj_kernel(const int* __restrict__ adjs, uint32_t* __restrict__ pm) {
  int e = blockIdx.x*256 + threadIdx.x;
  unsigned long long bal = __ballot(adjs[e] != 0);
  if ((threadIdx.x & 31) == 0) pm[e >> 5] = (uint32_t)bal;
}

// ---------------------------------------------------------------- gather
__global__ void gather_kernel(const float* __restrict__ vents,
                              const float* __restrict__ renc,
                              const int* __restrict__ rels,
                              float* __restrict__ xf, __bf16* __restrict__ xb) {
  int bn = blockIdx.x;          // 0..12799
  int c  = threadIdx.x;         // 0..127
  int b = bn / NN, n = bn % NN;
  float v;
  if (n < EE) v = vents[((size_t)b*EE + n)*HSZ + c];
  else {
    int r = rels[b*RR + (n - EE)];
    v = renc[(size_t)r*HSZ + c];
  }
  size_t o = (size_t)bn*HSZ + c;
  xf[o] = v;
  xb[o] = (__bf16)v;
}

// ---------------------------------------------------------------- weight prep (per layer)
// Pre-transpose all weights into bf16 [n][k] so GEMM tiles are pure async copies.
__global__ void prep_weights_kernel(const float* __restrict__ Wq, const float* __restrict__ Wk,
                                    const float* __restrict__ Wv, const float* __restrict__ Wo,
                                    const float* __restrict__ W1, const float* __restrict__ W2,
                                    __bf16* __restrict__ wqT, __bf16* __restrict__ wkT,
                                    __bf16* __restrict__ wvT, __bf16* __restrict__ woT,
                                    __bf16* __restrict__ w1T, __bf16* __restrict__ w2T) {
  int e = blockIdx.x*256 + threadIdx.x;       // 0..196607
  if (e < 65536) {                            // 4x 128x128
    int wm = e >> 14, i = e & 16383;
    int k = i >> 7, n = i & 127;
    const float* src = (wm==0)?Wq:(wm==1)?Wk:(wm==2)?Wv:Wo;
    __bf16* dst = (wm==0)?wqT:(wm==1)?wkT:(wm==2)?wvT:woT;
    dst[n*128 + k] = (__bf16)src[k*128 + n];
  } else if (e < 131072) {                    // W1: [128][512] -> w1T [512][128]
    int i = e - 65536;
    int n = i >> 7, k = i & 127;
    w1T[i] = (__bf16)W1[(size_t)k*512 + n];
  } else {                                    // W2: [512][128] -> w2T [128][512]
    int i = e - 131072;
    int n = i >> 9, k = i & 511;
    w2T[i] = (__bf16)W2[(size_t)k*128 + n];
  }
}

// ---------------------------------------------------------------- QKV GEMM
// grid: (100, 1, 3), block 256.  which = blockIdx.z : 0=Q (scaled), 1=K, 2=V (transposed out)
__global__ void qkv_kernel(const __bf16* __restrict__ xb,
                           const __bf16* __restrict__ wqT, const __bf16* __restrict__ wkT,
                           const __bf16* __restrict__ wvT,
                           const float* __restrict__ bq, const float* __restrict__ bk,
                           const float* __restrict__ bv,
                           __bf16* __restrict__ qb, __bf16* __restrict__ kb,
                           __bf16* __restrict__ vT) {
  __shared__ __align__(16) __bf16 Als[128*128];
  __shared__ __align__(16) __bf16 Wls[128*128];
  int tid = threadIdx.x;
  int which = blockIdx.z;
  int row0 = blockIdx.x * 128;
  const __bf16* WT = (which == 0) ? wqT : (which == 1) ? wkT : wvT;
  const float* bias = (which == 0) ? bq : (which == 1) ? bk : bv;

  async_copy_contig(lds_offset_of(Als), xb + (size_t)row0*HSZ);
  async_copy_contig(lds_offset_of(Wls), WT);
  async_wait_all();
  __syncthreads();

  int lane = tid & 31, wv = tid >> 5;
  v8f z = {0.f,0.f,0.f,0.f,0.f,0.f,0.f,0.f};
  v8f acc[8];
#pragma unroll
  for (int i = 0; i < 8; ++i) acc[i] = z;
#pragma unroll
  for (int kk = 0; kk < 128; kk += 32) {
    v16bf a = load_frag_a(Als + wv*16*128 + kk, 128);
#pragma unroll
    for (int nt = 0; nt < 8; ++nt)
      acc[nt] = wmma_bf16(a, load_frag_b(Wls + nt*16*128 + kk, 128), acc[nt]);
  }

  int rh = (lane >> 4) * 8, cl = lane & 15;
  const float qscale = 0.17677669529663687f;   // 1/sqrt(32)
#pragma unroll
  for (int nt = 0; nt < 8; ++nt) {
#pragma unroll
    for (int g = 0; g < 8; ++g) {
      int grow = row0 + wv*16 + rh + g;
      int col  = nt*16 + cl;
      float val = acc[nt][g] + bias[col];
      if (which == 0) {
        qb[(size_t)grow*HSZ + col] = (__bf16)(val * qscale);
      } else if (which == 1) {
        kb[(size_t)grow*HSZ + col] = (__bf16)val;
      } else {
        int b = grow / NN, n = grow % NN;
        int h = col >> 5, d = col & 31;
        vT[(((size_t)b*NH + h)*DKD + d)*NN + n] = (__bf16)val;
      }
    }
  }
}

// ---------------------------------------------------------------- flash attention
// grid: (13, NH, BB), block 256 (8 waves, one 16-query tile each)
__global__ void attn_kernel(const __bf16* __restrict__ qb, const __bf16* __restrict__ kb,
                            const __bf16* __restrict__ vT, const uint32_t* __restrict__ pm,
                            __bf16* __restrict__ ao) {
  __shared__ __align__(16) __bf16 pls[8][16][32];
  int lane = threadIdx.x & 31, wv = threadIdx.x >> 5;
  int qt = blockIdx.x * 8 + wv;
  if (qt >= NN/16) return;                    // wave-uniform
  int h = blockIdx.y, b = blockIdx.z;
  int q0 = qt * 16;

  v16bf aq = load_frag_a(qb + ((size_t)(b*NN + q0))*HSZ + h*DKD, HSZ);

  v8f z = {0.f,0.f,0.f,0.f,0.f,0.f,0.f,0.f};
  v8f acc0 = z, acc1 = z;
  float rmax[8], rsum[8];
#pragma unroll
  for (int g = 0; g < 8; ++g) { rmax[g] = -1e30f; rsum[g] = 0.f; }

  int rh = (lane >> 4) * 8, cl = lane & 15;
  const uint32_t* pmb = pm + (size_t)b*NN*NW;
  const __bf16* vbase = vT + ((size_t)(b*NH + h))*DKD*NN;

  for (int k0 = 0; k0 < NN; k0 += 32) {
    v8f s0 = wmma_bf16(aq, load_frag_b(kb + ((size_t)(b*NN + k0     ))*HSZ + h*DKD, HSZ), z);
    v8f s1 = wmma_bf16(aq, load_frag_b(kb + ((size_t)(b*NN + k0 + 16))*HSZ + h*DKD, HSZ), z);

#pragma unroll
    for (int g = 0; g < 8; ++g) {
      int r = rh + g;
      uint32_t mw = pmb[(q0 + r)*NW + (k0 >> 5)];          // bits for keys k0..k0+31
      float v0 = (mw & (1u << cl))        ? s0[g] : -1e9f;
      float v1 = (mw & (1u << (16 + cl))) ? s1[g] : -1e9f;
      float m = fmaxf(v0, v1);
      m = fmaxf(m, __shfl_xor(m, 1, 32));
      m = fmaxf(m, __shfl_xor(m, 2, 32));
      m = fmaxf(m, __shfl_xor(m, 4, 32));
      m = fmaxf(m, __shfl_xor(m, 8, 32));
      float nm = fmaxf(rmax[g], m);
      float alpha = __expf(rmax[g] - nm);
      float p0 = __expf(v0 - nm);
      float p1 = __expf(v1 - nm);
      float ps = p0 + p1;
      ps += __shfl_xor(ps, 1, 32);
      ps += __shfl_xor(ps, 2, 32);
      ps += __shfl_xor(ps, 4, 32);
      ps += __shfl_xor(ps, 8, 32);
      rsum[g] = rsum[g]*alpha + ps;
      rmax[g] = nm;
      acc0[g] *= alpha;
      acc1[g] *= alpha;
      pls[wv][r][cl]      = (__bf16)p0;
      pls[wv][r][16 + cl] = (__bf16)p1;
    }
    __asm__ volatile("" ::: "memory");   // order LDS p stores vs fragment reload (in-order per wave)

    v16bf ap = load_frag_a(&pls[wv][0][0], 32);
    acc0 = wmma_bf16(ap, load_frag_b(vbase +          0*NN + k0, NN), acc0);
    acc1 = wmma_bf16(ap, load_frag_b(vbase + (size_t)16*NN + k0, NN), acc1);
  }

  size_t outb = ((size_t)(b*NN + q0))*HSZ + h*DKD;
#pragma unroll
  for (int g = 0; g < 8; ++g) {
    int r = rh + g;
    float inv = 1.0f / rsum[g];
    ao[outb + (size_t)r*HSZ + cl]      = (__bf16)(acc0[g]*inv);
    ao[outb + (size_t)r*HSZ + 16 + cl] = (__bf16)(acc1[g]*inv);
  }
}

// ---------------------------------------------------------------- Wo GEMM + LayerNorm
// grid 100, block 256, dynamic LDS 64KB
__global__ void proj_ln_kernel(const __bf16* __restrict__ ain,
                               const __bf16* __restrict__ woT, const float* __restrict__ bo,
                               const float* __restrict__ g1, const float* __restrict__ be1,
                               float* __restrict__ tf, __bf16* __restrict__ tb) {
  extern __shared__ __align__(16) char smem[];
  __bf16* Als = (__bf16*)smem;
  __bf16* Wls = (__bf16*)(smem + 32768);
  float*  Ols = (float*)smem;                 // reused after compute
  int tid = threadIdx.x;
  int row0 = blockIdx.x * 128;

  async_copy_contig(lds_offset_of(Als), ain + (size_t)row0*HSZ);
  async_copy_contig(lds_offset_of(Wls), woT);
  async_wait_all();
  __syncthreads();

  int lane = tid & 31, wv = tid >> 5;
  v8f z = {0.f,0.f,0.f,0.f,0.f,0.f,0.f,0.f};
  v8f acc[8];
#pragma unroll
  for (int i = 0; i < 8; ++i) acc[i] = z;
#pragma unroll
  for (int kk = 0; kk < 128; kk += 32) {
    v16bf a = load_frag_a(Als + wv*16*128 + kk, 128);
#pragma unroll
    for (int nt = 0; nt < 8; ++nt)
      acc[nt] = wmma_bf16(a, load_frag_b(Wls + nt*16*128 + kk, 128), acc[nt]);
  }
  __syncthreads();

  int rh = (lane >> 4) * 8, cl = lane & 15;
#pragma unroll
  for (int nt = 0; nt < 8; ++nt)
#pragma unroll
    for (int g = 0; g < 8; ++g)
      Ols[(wv*16 + rh + g)*128 + nt*16 + cl] = acc[nt][g] + bo[nt*16 + cl];
  __syncthreads();

  if (tid < 128) {
    const float* rowp = Ols + tid*128;
    float mu = 0.f;
    for (int c = 0; c < 128; ++c) mu += rowp[c];
    mu *= (1.0f/128.0f);
    float var = 0.f;
    for (int c = 0; c < 128; ++c) { float d = rowp[c]-mu; var += d*d; }
    var *= (1.0f/128.0f);
    float inv = rsqrtf(var + 1e-5f);
    size_t ob = (size_t)(row0 + tid)*HSZ;
    for (int c = 0; c < 128; ++c) {
      float tv = (rowp[c]-mu)*inv*g1[c] + be1[c];
      tf[ob+c] = tv;
      tb[ob+c] = (__bf16)tv;
    }
  }
}

// ---------------------------------------------------------------- FFN1 (+gelu)
// grid (100,4), block 256
__global__ void ffn1_kernel(const __bf16* __restrict__ tb,
                            const __bf16* __restrict__ w1T, const float* __restrict__ b1,
                            __bf16* __restrict__ hb) {
  __shared__ __align__(16) __bf16 Als[128*128];
  __shared__ __align__(16) __bf16 Wls[128*128];
  int tid = threadIdx.x;
  int row0 = blockIdx.x * 128;
  int n0   = blockIdx.y * 128;

  async_copy_contig(lds_offset_of(Als), tb + (size_t)row0*HSZ);
  async_copy_contig(lds_offset_of(Wls), w1T + (size_t)n0*128);   // contiguous rows of [512][128]
  async_wait_all();
  __syncthreads();

  int lane = tid & 31, wv = tid >> 5;
  v8f z = {0.f,0.f,0.f,0.f,0.f,0.f,0.f,0.f};
  v8f acc[8];
#pragma unroll
  for (int i = 0; i < 8; ++i) acc[i] = z;
#pragma unroll
  for (int kk = 0; kk < 128; kk += 32) {
    v16bf a = load_frag_a(Als + wv*16*128 + kk, 128);
#pragma unroll
    for (int nt = 0; nt < 8; ++nt)
      acc[nt] = wmma_bf16(a, load_frag_b(Wls + nt*16*128 + kk, 128), acc[nt]);
  }

  int rh = (lane >> 4) * 8, cl = lane & 15;
#pragma unroll
  for (int nt = 0; nt < 8; ++nt) {
#pragma unroll
    for (int g = 0; g < 8; ++g) {
      int row = row0 + wv*16 + rh + g;
      int col = nt*16 + cl;
      float xv = acc[nt][g] + b1[n0 + col];
      float gl = 0.5f*xv*(1.0f + tanhf(0.7978845608028654f*(xv + 0.044715f*xv*xv*xv)));
      hb[(size_t)row*512 + n0 + col] = (__bf16)gl;
    }
  }
}

// ---------------------------------------------------------------- FFN2 + residual + double LN
// grid 100, block 256, dynamic LDS 64KB
__global__ void ffn2_kernel(const __bf16* __restrict__ hb,
                            const __bf16* __restrict__ w2T, const float* __restrict__ b2,
                            const float* __restrict__ g1, const float* __restrict__ be1,
                            const float* __restrict__ tf,
                            float* __restrict__ xf, __bf16* __restrict__ xb) {
  extern __shared__ __align__(16) char smem[];
  __bf16* Als = (__bf16*)smem;
  __bf16* Wls = (__bf16*)(smem + 32768);
  float*  Ols = (float*)smem;
  int tid = threadIdx.x;
  int row0 = blockIdx.x * 128;
  int lane = tid & 31, wv = tid >> 5;

  v8f z = {0.f,0.f,0.f,0.f,0.f,0.f,0.f,0.f};
  v8f acc[8];
#pragma unroll
  for (int i = 0; i < 8; ++i) acc[i] = z;

  for (int kc = 0; kc < 512; kc += 128) {
    if (kc) __syncthreads();
    // A chunk: rows of hb [12800][512], 256B per row, stride 1024B
    async_copy_rows(lds_offset_of(Als), hb + (size_t)row0*512 + kc, 1024);
    // W chunk: rows of w2T [128][512], 256B per row at col kc, stride 1024B
    async_copy_rows(lds_offset_of(Wls), w2T + kc, 1024);
    async_wait_all();
    __syncthreads();
#pragma unroll
    for (int kk = 0; kk < 128; kk += 32) {
      v16bf a = load_frag_a(Als + wv*16*128 + kk, 128);
#pragma unroll
      for (int nt = 0; nt < 8; ++nt)
        acc[nt] = wmma_bf16(a, load_frag_b(Wls + nt*16*128 + kk, 128), acc[nt]);
    }
  }
  __syncthreads();

  int rh = (lane >> 4) * 8, cl = lane & 15;
#pragma unroll
  for (int nt = 0; nt < 8; ++nt)
#pragma unroll
    for (int g = 0; g < 8; ++g) {
      int row = wv*16 + rh + g;
      int col = nt*16 + cl;
      Ols[row*128 + col] = acc[nt][g] + b2[col] + tf[(size_t)(row0 + row)*HSZ + col];
    }
  __syncthreads();

  if (tid < 128) {
    const float* rowp = Ols + tid*128;
    float mu1 = 0.f;
    for (int c = 0; c < 128; ++c) mu1 += rowp[c];
    mu1 *= (1.0f/128.0f);
    float var1 = 0.f;
    for (int c = 0; c < 128; ++c) { float d = rowp[c]-mu1; var1 += d*d; }
    var1 *= (1.0f/128.0f);
    float inv1 = rsqrtf(var1 + 1e-5f);
    float mu2 = 0.f;
    for (int c = 0; c < 128; ++c) mu2 += (rowp[c]-mu1)*inv1*g1[c] + be1[c];
    mu2 *= (1.0f/128.0f);
    float var2 = 0.f;
    for (int c = 0; c < 128; ++c) {
      float zz = (rowp[c]-mu1)*inv1*g1[c] + be1[c];
      float d = zz - mu2; var2 += d*d;
    }
    var2 *= (1.0f/128.0f);
    float inv2 = rsqrtf(var2 + 1e-5f);
    size_t ob = (size_t)(row0 + tid)*HSZ;
    for (int c = 0; c < 128; ++c) {
      float zz = (rowp[c]-mu1)*inv1*g1[c] + be1[c];
      float fv = (zz - mu2)*inv2;
      xf[ob+c] = fv;
      xb[ob+c] = (__bf16)fv;
    }
  }
}

// ---------------------------------------------------------------- output scatter
__global__ void writeout_kernel(const float* __restrict__ x, float* __restrict__ out) {
  int bn = blockIdx.x, c = threadIdx.x;
  int b = bn / NN, n = bn % NN;
  float v = x[(size_t)bn*HSZ + c];
  if (n < EE)       out[((size_t)b*EE + n)*HSZ + c] = v;                         // gents
  else if (n == EE) out[825600 + b*HSZ + c] = v;                                 // glob
  else              out[826624 + ((size_t)b*(RR-1) + (n - EE - 1))*HSZ + c] = v; // grels
}

__global__ void mask_kernel(float* __restrict__ out) {
  int i = blockIdx.x*256 + threadIdx.x;
  if (i < BB*EE)     out[819200  + i] = 1.0f;   // emask
  if (i < BB*(RR-1)) out[1644800 + i] = 1.0f;   // rmask
}

// ---------------------------------------------------------------- launcher
extern "C" void kernel_launch(void* const* d_in, const int* in_sizes, int n_in,
                              void* d_out, int out_size, void* d_ws, size_t ws_size,
                              hipStream_t stream) {
  const int*   adjs  = (const int*)d_in[0];
  const int*   rels  = (const int*)d_in[1];
  const float* vents = (const float*)d_in[2];
  const float* renc  = (const float*)d_in[3];
  const float* Wq = (const float*)d_in[4];
  const float* bq = (const float*)d_in[5];
  const float* Wk = (const float*)d_in[6];
  const float* bk = (const float*)d_in[7];
  const float* Wv = (const float*)d_in[8];
  const float* bv = (const float*)d_in[9];
  const float* Wo = (const float*)d_in[10];
  const float* bo = (const float*)d_in[11];
  const float* W1 = (const float*)d_in[12];
  const float* b1 = (const float*)d_in[13];
  const float* W2 = (const float*)d_in[14];
  const float* b2 = (const float*)d_in[15];
  const float* g1 = (const float*)d_in[16];
  const float* be1= (const float*)d_in[17];

  char* ws = (char*)d_ws;
  size_t off = 0;
  auto alloc = [&](size_t bytes) {
    void* p = ws + off;
    off += (bytes + 255) & ~(size_t)255;
    return p;
  };
  float*    x_f32 = (float*)   alloc((size_t)MROWS*HSZ*4);
  __bf16*   x_bf  = (__bf16*)  alloc((size_t)MROWS*HSZ*2);
  __bf16*   qb    = (__bf16*)  alloc((size_t)MROWS*HSZ*2);
  __bf16*   kb    = (__bf16*)  alloc((size_t)MROWS*HSZ*2);
  __bf16*   vT    = (__bf16*)  alloc((size_t)MROWS*HSZ*2);
  __bf16*   ao    = (__bf16*)  alloc((size_t)MROWS*HSZ*2);
  float*    t_f32 = (float*)   alloc((size_t)MROWS*HSZ*4);
  __bf16*   t_bf  = (__bf16*)  alloc((size_t)MROWS*HSZ*2);
  __bf16*   hb    = (__bf16*)  alloc((size_t)MROWS*512*2);
  uint32_t* pm    = (uint32_t*)alloc((size_t)BB*NN*NW*4);      // 2.56 MB packed mask
  __bf16*   wqT   = (__bf16*)  alloc(16384*2);
  __bf16*   wkT   = (__bf16*)  alloc(16384*2);
  __bf16*   wvT   = (__bf16*)  alloc(16384*2);
  __bf16*   woT   = (__bf16*)  alloc(16384*2);
  __bf16*   w1T   = (__bf16*)  alloc(65536*2);
  __bf16*   w2T   = (__bf16*)  alloc(65536*2);

  pack_adj_kernel<<<80000, 256, 0, stream>>>(adjs, pm);
  gather_kernel<<<MROWS, 128, 0, stream>>>(vents, renc, rels, x_f32, x_bf);

  for (int j = 0; j < 2; ++j) {
    prep_weights_kernel<<<768, 256, 0, stream>>>(
        Wq + j*128*128, Wk + j*128*128, Wv + j*128*128, Wo + j*128*128,
        W1 + j*128*512, W2 + j*512*128, wqT, wkT, wvT, woT, w1T, w2T);
    qkv_kernel<<<dim3(100,1,3), 256, 0, stream>>>(
        x_bf, wqT, wkT, wvT, bq + j*128, bk + j*128, bv + j*128, qb, kb, vT);
    attn_kernel<<<dim3(13, NH, BB), 256, 0, stream>>>(qb, kb, vT, pm, ao);
    proj_ln_kernel<<<dim3(100), 256, 65536, stream>>>(
        ao, woT, bo + j*128, g1 + j*128, be1 + j*128, t_f32, t_bf);
    ffn1_kernel<<<dim3(100,4), 256, 0, stream>>>(t_bf, w1T, b1 + j*512, hb);
    ffn2_kernel<<<dim3(100), 256, 65536, stream>>>(
        hb, w2T, b2 + j*128, g1 + j*128, be1 + j*128, t_f32, x_f32, x_bf);
  }

  writeout_kernel<<<MROWS, 128, 0, stream>>>(x_f32, (float*)d_out);
  mask_kernel<<<25, 256, 0, stream>>>((float*)d_out);
}
